// MPNN_edge_22771916604015
// MI455X (gfx1250) — matrix operand order repaired and checked by reading, not compiled, not run on hardware
//
#include <hip/hip_runtime.h>

typedef _Float16 v16h __attribute__((ext_vector_type(16)));
typedef _Float16 v8h  __attribute__((ext_vector_type(8)));
typedef _Float16 f16x2 __attribute__((ext_vector_type(2)));
typedef float    v8f  __attribute__((ext_vector_type(8)));

#define Bsz 4
#define Nsz 256
#define Fsz 64
#define Esz 64
#define ZP  200                    // z-buffer pitch in halves (16B-aligned rows, conflict-free)
#define WAVES_PER_BLOCK 8
#define THREADS (WAVES_PER_BLOCK * 32)
#define GRID 512
#define TILES (Bsz * Nsz * Nsz / 16)   // 16384 tiles of 16 rows
#define TILE_STRIDE (GRID * WAVES_PER_BLOCK)

#define FRAG_H 512                 // halves per B fragment (32 lanes x 16 halves)
// fragment counts per matrix: KT*NT
#define FR_MW0 (6*8)
#define FR_MW1 (4*8)
#define FR_MW4 (4*4)
#define FR_WIH (2*12)
// LDS half-offsets of packed weights
#define OFF_MW0 0
#define OFF_MW1 (OFF_MW0 + FR_MW0*FRAG_H)
#define OFF_MW2 (OFF_MW1 + FR_MW1*FRAG_H)
#define OFF_MW3 (OFF_MW2 + FR_MW1*FRAG_H)
#define OFF_MW4 (OFF_MW3 + FR_MW1*FRAG_H)
#define OFF_WIH (OFF_MW4 + FR_MW4*FRAG_H)
#define OFF_WHH (OFF_WIH + FR_WIH*FRAG_H)
#define W_HALVES (OFF_WHH + FR_WIH*FRAG_H)     // 106496 halves = 212992 B
#define BIAS_FLOATS 960                          // mb0..mb3(128ea) mb4(64) b_ih(192) b_hh(192)
#define SMEM_BYTES ((size_t)W_HALVES*2 + BIAS_FLOATS*4 + (size_t)WAVES_PER_BLOCK*16*ZP*2)

// ---- helpers -------------------------------------------------------------

__device__ __forceinline__ v16h cat16(v8h a, v8h b) {
    return __builtin_shufflevector(a, b, 0,1,2,3,4,5,6,7,8,9,10,11,12,13,14,15);
}

// B fragment: packed contiguously per lane at pack time -> 2x ds_load_b128
__device__ __forceinline__ v16h ld_bfrag(const _Float16* wf, int frag, int lane) {
    const _Float16* p = wf + frag * FRAG_H + lane * 16;
    return cat16(*(const v8h*)p, *(const v8h*)(p + 8));
}

// A fragment per ISA 16-bit A 16x32 layout:
// lanes 0-15 (kh=0): K = [kb, kb+8) then [kb+16, kb+24)
// lanes 16-31(kh=1): K = [kb+8, kb+16) then [kb+24, kb+32)
__device__ __forceinline__ v16h ld_afrag(const _Float16* zrow, int kb, int kh) {
    v8h c0 = *(const v8h*)(zrow + kb + 8*kh);
    v8h c1 = *(const v8h*)(zrow + kb + 16 + 8*kh);
    return cat16(c0, c1);
}

// Cooperative pack: global f32 weights -> per-lane-contiguous f16 B fragments in LDS.
// B[k][n] = transp ? W[n*K+k] : W[k*Nout+n]; lane holds column n=16*nt+(l&15),
// halves e=0..15 -> K = 32*ks + 8*(l>>4) + (e&7) + 16*(e>>3).
__device__ void pack_weights(const float* __restrict__ W, int K, int Nout, bool transp,
                             _Float16* dst) {
    int NT = Nout >> 4;
    int total = (K >> 5) * NT * FRAG_H;
    for (int h = threadIdx.x; h < total; h += THREADS) {
        int frag = h >> 9;
        int within = h & 511;
        int lane = within >> 4;
        int e = within & 15;
        int ks = frag / NT, nt = frag - ks * NT;
        int k = 32*ks + 8*(lane >> 4) + (e & 7) + 16*(e >> 3);
        int n = 16*nt + (lane & 15);
        float v = transp ? W[n * K + k] : W[k * Nout + n];
        dst[h] = (_Float16)v;
    }
}

// One GEMM layer: [16 x 32*KT] (z-buffer cols acol..) @ packed weights -> acc[NT].
// B fragments double-buffered over n-tile PAIRS: next pair's 2*KT ds loads are
// issued before the current pair's WMMAs, so the wait is dscnt<=2*KT (overlap),
// and the 2*KT WMMAs form two independent interleaved accumulation chains.
template<int KT, int NT>
__device__ __forceinline__ void do_layer(const _Float16* zbuf, int acol,
                                         const _Float16* wf, v8f* acc, int lane) {
    static_assert((NT & 1) == 0, "NT must be even");
    constexpr int NP = NT / 2;
    int m = lane & 15, kh = lane >> 4;
    const _Float16* zrow = zbuf + m * ZP;
    v16h a[KT];
#pragma unroll
    for (int ks = 0; ks < KT; ++ks) a[ks] = ld_afrag(zrow, acol + 32*ks, kh);

    v16h b[2][2 * KT];
#pragma unroll
    for (int ks = 0; ks < KT; ++ks) {               // preload pair 0
        b[0][ks]      = ld_bfrag(wf, ks * NT + 0, lane);
        b[0][KT + ks] = ld_bfrag(wf, ks * NT + 1, lane);
    }
#pragma unroll
    for (int p = 0; p < NP; ++p) {
        const int cur = p & 1;
        if (p + 1 < NP) {                           // prefetch pair p+1
            const int nxt = (p + 1) & 1;
            const int nt = 2 * (p + 1);
#pragma unroll
            for (int ks = 0; ks < KT; ++ks) {
                b[nxt][ks]      = ld_bfrag(wf, ks * NT + nt,     lane);
                b[nxt][KT + ks] = ld_bfrag(wf, ks * NT + nt + 1, lane);
            }
        }
        v8f c0 = {}, c1 = {};
#pragma unroll
        for (int ks = 0; ks < KT; ++ks) {           // two independent chains
            c0 = __builtin_amdgcn_wmma_f32_16x16x32_f16(false, a[ks], false, b[cur][ks],
                                                        (short)0, c0, false, false);
            c1 = __builtin_amdgcn_wmma_f32_16x16x32_f16(false, a[ks], false, b[cur][KT + ks],
                                                        (short)0, c1, false, false);
        }
        acc[2 * p]     = c0;
        acc[2 * p + 1] = c1;
    }
}

// bias (+ReLU) epilogue; scatter D-layout accumulators back to z-buffer as f16
template<int NT, bool RELU>
__device__ __forceinline__ void store_act(_Float16* zbuf, const v8f* acc,
                                          const float* bias, int lane) {
    int c = lane & 15, kh = lane >> 4;
#pragma unroll
    for (int nt = 0; nt < NT; ++nt) {
        float bv = bias[16*nt + c];
#pragma unroll
        for (int v = 0; v < 8; ++v) {
            float x = acc[nt][v] + bv;
            if (RELU) x = fmaxf(x, 0.0f);
            zbuf[(v + 8*kh) * ZP + 16*nt + c] = (_Float16)x;
        }
    }
}

// ---- kernel --------------------------------------------------------------

extern "C" __global__ __launch_bounds__(THREADS, 1)
void mpnn_edge_kernel(const float* __restrict__ edge, const float* __restrict__ x,
                      const float* __restrict__ mw0, const float* __restrict__ mb0,
                      const float* __restrict__ mw1, const float* __restrict__ mb1,
                      const float* __restrict__ mw2, const float* __restrict__ mb2,
                      const float* __restrict__ mw3, const float* __restrict__ mb3,
                      const float* __restrict__ mw4, const float* __restrict__ mb4,
                      const float* __restrict__ wih, const float* __restrict__ whh,
                      const float* __restrict__ bih, const float* __restrict__ bhh,
                      float* __restrict__ out) {
    extern __shared__ char smem_raw[];
    _Float16* wh    = (_Float16*)smem_raw;
    float*    bias  = (float*)(smem_raw + (size_t)W_HALVES * 2);
    _Float16* zbase = (_Float16*)(smem_raw + (size_t)W_HALVES * 2 + BIAS_FLOATS * 4);

    // ---- one-time per block: pack weights (f32->f16, fragment-major) ----
    pack_weights(mw0, 192, 128, false, wh + OFF_MW0);
    pack_weights(mw1, 128, 128, false, wh + OFF_MW1);
    pack_weights(mw2, 128, 128, false, wh + OFF_MW2);
    pack_weights(mw3, 128, 128, false, wh + OFF_MW3);
    pack_weights(mw4, 128,  64, false, wh + OFF_MW4);
    pack_weights(wih,  64, 192, true,  wh + OFF_WIH);  // B = w_ih^T (64x192)
    pack_weights(whh,  64, 192, true,  wh + OFF_WHH);  // B = w_hh^T (64x192)
    for (int idx = threadIdx.x; idx < BIAS_FLOATS; idx += THREADS) {
        float v;
        if      (idx < 128) v = mb0[idx];
        else if (idx < 256) v = mb1[idx - 128];
        else if (idx < 384) v = mb2[idx - 256];
        else if (idx < 512) v = mb3[idx - 384];
        else if (idx < 576) v = mb4[idx - 512];
        else if (idx < 768) v = bih[idx - 576];
        else                v = bhh[idx - 768];
        bias[idx] = v;
    }
    __syncthreads();

    const int lane = threadIdx.x & 31;
    const int wid  = threadIdx.x >> 5;
    const int c    = lane & 15;
    const int kh   = lane >> 4;
    _Float16* zbuf = zbase + wid * (16 * ZP);
    const int gw   = blockIdx.x * WAVES_PER_BLOCK + wid;

    // wave-uniform tile loop (16384 tiles / 4096 waves = 4 each) -> EXEC all-1 at WMMA
    for (int tile = gw; tile < TILES; tile += TILE_STRIDE) {
        int rb  = tile << 4;           // flat row base in [B*N*N)
        int b   = rb >> 16;            // / (N*N)
        int rem = rb & 65535;
        int i   = rem >> 8;            // / N
        int j0  = rem & 255;           // % N (multiple of 16)
        const float* xb = x + (size_t)b * Nsz * Fsz;
        const float* eg = edge + ((size_t)(b * Nsz + i) * Nsz + j0) * Esz;
        float*       og = out  + ((size_t)(b * Nsz + i) * Nsz + j0) * Esz;

        // prefetch next tile's edge block (16x64 f32 = 4KB; one 128B line per lane)
        {
            int tn = tile + TILE_STRIDE;
            if (tn < TILES) {
                int rbn = tn << 4;
                const float* egn = edge + (size_t)rbn * Esz;
                __builtin_prefetch(egn + lane * 32, 0, 3);
            }
        }

        // ---- fill z-buffer [16 x 192] f16: [ x_j | x_i | edge ] ----
        for (int p = lane; p < 512; p += 32) {           // h1: x[b, j0+m, :]
            int m = p >> 5, cp = (p & 31) * 2;
            const float* s = xb + (j0 + m) * Fsz + cp;
            f16x2 hv = { (_Float16)s[0], (_Float16)s[1] };
            *(f16x2*)(zbuf + m * ZP + cp) = hv;
        }
        {                                                // h2: x[b, i, :] broadcast
            int cp = lane * 2;
            const float* s = xb + i * Fsz + cp;
            f16x2 hv = { (_Float16)s[0], (_Float16)s[1] };
#pragma unroll
            for (int m = 0; m < 16; ++m) *(f16x2*)(zbuf + m * ZP + 64 + cp) = hv;
        }
        for (int p = lane; p < 512; p += 32) {           // edge tile (cols 128..191)
            int m = p >> 5, cp = (p & 31) * 2;
            const float* s = eg + m * Esz + cp;
            f16x2 hv = { (_Float16)s[0], (_Float16)s[1] };
            *(f16x2*)(zbuf + m * ZP + 128 + cp) = hv;
        }

        // ---- MLP: 192->128->128->128->128->64 (edge f16 at cols 128..191 survives) ----
        v8f acc[8];
        do_layer<6, 8>(zbuf, 0, wh + OFF_MW0, acc, lane);
        store_act<8, true >(zbuf, acc, bias + 0,   lane);
        do_layer<4, 8>(zbuf, 0, wh + OFF_MW1, acc, lane);
        store_act<8, true >(zbuf, acc, bias + 128, lane);
        do_layer<4, 8>(zbuf, 0, wh + OFF_MW2, acc, lane);
        store_act<8, true >(zbuf, acc, bias + 256, lane);
        do_layer<4, 8>(zbuf, 0, wh + OFF_MW3, acc, lane);
        store_act<8, true >(zbuf, acc, bias + 384, lane);
        do_layer<4, 4>(zbuf, 0, wh + OFF_MW4, acc, lane);
        store_act<4, false>(zbuf, acc, bias + 512, lane);   // message -> cols 0..63

        // ---- GRU: gi = msg @ w_ih^T, gh = edge @ w_hh^T (both 16x192) ----
        v8f gi[12], gh[12];
        do_layer<2, 12>(zbuf, 0,   wh + OFF_WIH, gi, lane);
        do_layer<2, 12>(zbuf, 128, wh + OFF_WHH, gh, lane);

        const float* bih_l = bias + 576;
        const float* bhh_l = bias + 768;
#pragma unroll
        for (int t = 0; t < 4; ++t) {
            float bir = bih_l[16*t + c], biz = bih_l[64 + 16*t + c], bin = bih_l[128 + 16*t + c];
            float bhr = bhh_l[16*t + c], bhz = bhh_l[64 + 16*t + c], bhn = bhh_l[128 + 16*t + c];
#pragma unroll
            for (int v = 0; v < 8; ++v) {
                int m = v + 8 * kh;                      // D-layout row
                float hf = eg[m * Esz + 16*t + c];       // hidden = edge (f32)
                float ir = gi[t][v] + bir, iz = gi[t+4][v] + biz, in = gi[t+8][v] + bin;
                float hr = gh[t][v] + bhr, hz = gh[t+4][v] + bhz, hn = gh[t+8][v] + bhn;
                float r  = 1.0f / (1.0f + __expf(-(ir + hr)));
                float zg = 1.0f / (1.0f + __expf(-(iz + hz)));
                float ng = tanhf(in + r * hn);
                og[m * Esz + 16*t + c] = (1.0f - zg) * ng + zg * hf;
            }
        }
    }
}

// ---- host launch ---------------------------------------------------------

extern "C" void kernel_launch(void* const* d_in, const int* in_sizes, int n_in,
                              void* d_out, int out_size, void* d_ws, size_t ws_size,
                              hipStream_t stream) {
    const float* edge = (const float*)d_in[0];
    const float* x    = (const float*)d_in[1];
    const float* mw0  = (const float*)d_in[2];
    const float* mb0  = (const float*)d_in[3];
    const float* mw1  = (const float*)d_in[4];
    const float* mb1  = (const float*)d_in[5];
    const float* mw2  = (const float*)d_in[6];
    const float* mb2  = (const float*)d_in[7];
    const float* mw3  = (const float*)d_in[8];
    const float* mb3  = (const float*)d_in[9];
    const float* mw4  = (const float*)d_in[10];
    const float* mb4  = (const float*)d_in[11];
    const float* wih  = (const float*)d_in[12];
    const float* whh  = (const float*)d_in[13];
    const float* bih  = (const float*)d_in[14];
    const float* bhh  = (const float*)d_in[15];
    float* out = (float*)d_out;

    (void)in_sizes; (void)n_in; (void)out_size; (void)d_ws; (void)ws_size;

    // 268 KB dynamic LDS (<= 320 KB WGP limit); opt in every call (deterministic).
    hipFuncSetAttribute((const void*)mpnn_edge_kernel,
                        hipFuncAttributeMaxDynamicSharedMemorySize, (int)SMEM_BYTES);
    mpnn_edge_kernel<<<GRID, THREADS, SMEM_BYTES, stream>>>(
        edge, x, mw0, mb0, mw1, mb1, mw2, mb2, mw3, mb3, mw4, mb4,
        wih, whh, bih, bhh, out);
}
